// Gat_44667659878704
// MI455X (gfx1250) — matrix operand order repaired and checked by reading, not compiled, not run on hardware
//
#include <hip/hip_runtime.h>
#include <hip/hip_bf16.h>

// Problem constants (from reference)
#define D_DIM   512
#define KNB     32
#define HEADS   4
#define BN_TOT  4096          // B*N = 4*1024
#define NEGV    (-1.0e9f)
#define EPITCH  520           // LDS pitch (bf16 elems) for E tile: 512 + 8 (16B) to break bank conflicts

typedef __bf16 bf16_t;
typedef __attribute__((ext_vector_type(16))) __bf16 v16bf;
typedef __attribute__((ext_vector_type(8)))  __bf16 v8bf;
typedef __attribute__((ext_vector_type(8)))  float  v8f;
typedef __attribute__((ext_vector_type(4)))  float  v4f;

static __device__ __forceinline__ bf16_t f2bf(float f) {
    unsigned u = __builtin_bit_cast(unsigned, f);
    unsigned short r = (unsigned short)((u + 0x7FFFu + ((u >> 16) & 1u)) >> 16); // RNE
    return __builtin_bit_cast(bf16_t, r);
}

static __device__ __forceinline__ v16bf cat8(v8bf lo, v8bf hi) {
    return __builtin_shufflevector(lo, hi, 0,1,2,3,4,5,6,7,8,9,10,11,12,13,14,15);
}

static __device__ __forceinline__ v8f wmma_bf16(v16bf a, v16bf b, v8f c) {
    return __builtin_amdgcn_wmma_f32_16x16x32_bf16(false, a, false, b, (short)0, c, false, false);
}

// ---------------------------------------------------------------------------
// Kernel 1: fp32 -> bf16 conversion of Wq, Wk, Wv, x into workspace
// ---------------------------------------------------------------------------
__global__ __launch_bounds__(256) void cvt_kernel(
    const float* __restrict__ Wq, const float* __restrict__ Wk,
    const float* __restrict__ Wv, const float* __restrict__ x,
    bf16_t* __restrict__ Wq_bf, bf16_t* __restrict__ Wk_bf,
    bf16_t* __restrict__ Wv_bf, bf16_t* __restrict__ x_bf)
{
    const int DD = D_DIM * D_DIM;          // 262144
    const int NX = BN_TOT * D_DIM;         // 2097152
    // Vectorized: each thread handles 4 consecutive floats -> 4 bf16 (8B store).
    int i4 = (blockIdx.x * 256 + threadIdx.x) * 4;
    const float* src; bf16_t* dst; int off;
    if (i4 < DD)              { src = Wq; dst = Wq_bf; off = i4; }
    else if (i4 < 2 * DD)     { src = Wk; dst = Wk_bf; off = i4 - DD; }
    else if (i4 < 3 * DD)     { src = Wv; dst = Wv_bf; off = i4 - 2 * DD; }
    else                      { int j = i4 - 3 * DD; if (j >= NX) return;
                                src = x;  dst = x_bf;  off = j; }
    v4f f = *(const v4f*)(src + off);
    __attribute__((ext_vector_type(4))) __bf16 p;
    p[0] = f2bf(f[0]); p[1] = f2bf(f[1]); p[2] = f2bf(f[2]); p[3] = f2bf(f[3]);
    *(__attribute__((ext_vector_type(4))) __bf16*)(dst + off) = p;
}

// ---------------------------------------------------------------------------
// Kernel 2: q = x @ Wq^T + bq  (M=4096, N=512, K=512), bf16 WMMA, fp32 out
// Block = 256 threads (8 waves); each block does a 32-row M tile.
// Wave w owns columns [w*64, w*64+64): 4 N-tiles x 2 M-tiles of 16x16.
// ---------------------------------------------------------------------------
__global__ __launch_bounds__(256) void q_gemm(
    const bf16_t* __restrict__ x_bf, const bf16_t* __restrict__ Wq_bf,
    const float* __restrict__ bq, float* __restrict__ q_out)
{
    const int tid   = threadIdx.x;
    const int wave  = tid >> 5;
    const int lane  = tid & 31;
    const int lhalf = lane >> 4;
    const int l16   = lane & 15;
    const int mbase = blockIdx.x * 32;
    const int nbase = wave * 64;

    const v8f z8 = {0.f,0.f,0.f,0.f,0.f,0.f,0.f,0.f};
    v8f acc[4][2];
    #pragma unroll
    for (int nt = 0; nt < 4; ++nt) { acc[nt][0] = z8; acc[nt][1] = z8; }

    for (int kt = 0; kt < 16; ++kt) {
        const int k0 = kt * 32;
        v16bf a[2];
        #pragma unroll
        for (int mt = 0; mt < 2; ++mt) {
            const bf16_t* ap = x_bf + (size_t)(mbase + mt * 16 + l16) * D_DIM + k0 + lhalf * 16;
            a[mt] = cat8(*(const v8bf*)ap, *(const v8bf*)(ap + 8));
        }
        #pragma unroll
        for (int nt = 0; nt < 4; ++nt) {
            const int n = nbase + nt * 16 + l16;
            const bf16_t* wp = Wq_bf + (size_t)n * D_DIM + k0 + lhalf * 16;
            v16bf bfrag = cat8(*(const v8bf*)wp, *(const v8bf*)(wp + 8));
            #pragma unroll
            for (int mt = 0; mt < 2; ++mt)
                acc[nt][mt] = wmma_bf16(a[mt], bfrag, acc[nt][mt]);
        }
    }

    #pragma unroll
    for (int nt = 0; nt < 4; ++nt) {
        const int n  = nbase + nt * 16 + l16;
        const float bqn = bq[n];
        #pragma unroll
        for (int mt = 0; mt < 2; ++mt)
            #pragma unroll
            for (int r = 0; r < 8; ++r) {
                const int row = mbase + mt * 16 + r + 8 * lhalf;
                q_out[(size_t)row * D_DIM + n] = acc[nt][mt][r] + bqn;
            }
    }
}

// ---------------------------------------------------------------------------
// Kernel 3: fused per-(b,n) GAT attention.
// One 256-thread block per (b,n). E tile -> bf16 LDS (vectorized b128 stage);
// K and V GEMMs fused in one k-loop (accumulators in registers); scores
// reduced by shuffles + ds_add_f32; masked softmax; attn @ V folded from V
// accumulators.
// ---------------------------------------------------------------------------
__global__ __launch_bounds__(256) void gat_main(
    const float* __restrict__ neib, const int* __restrict__ mask,
    const bf16_t* __restrict__ Wk_bf, const float* __restrict__ bk,
    const bf16_t* __restrict__ Wv_bf, const float* __restrict__ bv,
    const float* __restrict__ q_in, float* __restrict__ out)
{
    __shared__ bf16_t Ebf[KNB * EPITCH];       // 33,280 B
    __shared__ float  qv[D_DIM];               //  2,048 B
    __shared__ float  sc[HEADS * KNB];         //    512 B
    __shared__ float  attn[HEADS * KNB];       //    512 B
    __shared__ float  mskf[KNB];               //    128 B

    const int bn  = blockIdx.x;
    const int tid = threadIdx.x;

    // Stage E tile: 32x512 fp32 -> bf16 LDS. 8 floats per step per thread:
    // two global b128 loads, 8 cvts, one ds_store_b128.
    const float* Ep = neib + (size_t)bn * KNB * D_DIM;
    #pragma unroll
    for (int g = 0; g < (KNB * D_DIM) / (256 * 8); ++g) {       // 8 steps
        const int e0  = (g * 256 + tid) * 8;
        const int row = e0 >> 9;
        const int col = e0 & 511;
        v4f f0 = *(const v4f*)(Ep + e0);
        v4f f1 = *(const v4f*)(Ep + e0 + 4);
        v8bf p;
        p[0] = f2bf(f0[0]); p[1] = f2bf(f0[1]); p[2] = f2bf(f0[2]); p[3] = f2bf(f0[3]);
        p[4] = f2bf(f1[0]); p[5] = f2bf(f1[1]); p[6] = f2bf(f1[2]); p[7] = f2bf(f1[3]);
        *(v8bf*)&Ebf[row * EPITCH + col] = p;                   // 16B aligned
    }
    // q row (fp32, already has bq): 512 floats via float4, threads 0..127.
    const float* qp = q_in + (size_t)bn * D_DIM;
    if (tid < 128) *(v4f*)&qv[tid * 4] = *(const v4f*)(qp + tid * 4);
    if (tid < HEADS * KNB) sc[tid] = 0.f;
    if (tid < KNB) mskf[tid] = (mask[(size_t)bn * KNB + tid] == 0) ? 1.f : 0.f;
    __syncthreads();

    const int wave  = tid >> 5;
    const int lane  = tid & 31;
    const int lhalf = lane >> 4;
    const int l16   = lane & 15;
    const int nbase = wave * 64;               // wave owns cols [nbase, nbase+64)
    const int h     = wave >> 1;               // head of those columns

    const v8f z8 = {0.f,0.f,0.f,0.f,0.f,0.f,0.f,0.f};
    v8f acck[4][2], accv[4][2];
    #pragma unroll
    for (int nt = 0; nt < 4; ++nt) {
        acck[nt][0] = z8; acck[nt][1] = z8;
        accv[nt][0] = z8; accv[nt][1] = z8;
    }

    // Fused K/V GEMM: [32 x 512] = Ebf(32x512) @ W^T, 16 k-steps of depth 32.
    for (int kt = 0; kt < 16; ++kt) {
        const int k0 = kt * 32;
        v16bf a[2];
        #pragma unroll
        for (int mt = 0; mt < 2; ++mt) {
            const bf16_t* ap = &Ebf[(mt * 16 + l16) * EPITCH + k0 + lhalf * 16];
            a[mt] = cat8(*(const v8bf*)ap, *(const v8bf*)(ap + 8));
        }
        #pragma unroll
        for (int nt = 0; nt < 4; ++nt) {
            const int n = nbase + nt * 16 + l16;
            const bf16_t* wkp = Wk_bf + (size_t)n * D_DIM + k0 + lhalf * 16;
            const bf16_t* wvp = Wv_bf + (size_t)n * D_DIM + k0 + lhalf * 16;
            v16bf bK = cat8(*(const v8bf*)wkp, *(const v8bf*)(wkp + 8));
            v16bf bV = cat8(*(const v8bf*)wvp, *(const v8bf*)(wvp + 8));
            #pragma unroll
            for (int mt = 0; mt < 2; ++mt) {
                acck[nt][mt] = wmma_bf16(a[mt], bK, acck[nt][mt]);
                accv[nt][mt] = wmma_bf16(a[mt], bV, accv[nt][mt]);
            }
        }
    }

    // Scores: sc[h][j] += sum_d q[d] * (K[j][d] + bk[d]) over this wave's cols.
    // C/D layout: lane element r = C[mt*16 + r + 8*lhalf][nbase + nt*16 + l16].
    #pragma unroll
    for (int nt = 0; nt < 4; ++nt) {
        const int n   = nbase + nt * 16 + l16;
        const float qn  = qv[n];
        const float bkn = bk[n];
        #pragma unroll
        for (int mt = 0; mt < 2; ++mt)
            #pragma unroll
            for (int r = 0; r < 8; ++r) {
                float c = (acck[nt][mt][r] + bkn) * qn;
                c += __shfl_xor(c, 1);
                c += __shfl_xor(c, 2);
                c += __shfl_xor(c, 4);
                c += __shfl_xor(c, 8);      // sum over 16 columns (same j)
                if (l16 == 0)
                    atomicAdd(&sc[h * KNB + mt * 16 + r + 8 * lhalf], c);
            }
    }
    __syncthreads();

    // Masked softmax over the 32 neighbors, one thread per head.
    if (tid < HEADS) {
        const int hh = tid;
        float mx = -3.4e38f;
        float s[KNB];
        #pragma unroll
        for (int j = 0; j < KNB; ++j) {
            s[j] = (mskf[j] != 0.f) ? NEGV : sc[hh * KNB + j];
            mx = fmaxf(mx, s[j]);
        }
        float sum = 0.f;
        #pragma unroll
        for (int j = 0; j < KNB; ++j) { s[j] = __expf(s[j] - mx); sum += s[j]; }
        const float inv = 1.f / sum;
        #pragma unroll
        for (int j = 0; j < KNB; ++j) attn[hh * KNB + j] = s[j] * inv;
    }
    __syncthreads();

    // vec[d] = sum_j attn[h][j] * V[j][d] + bv[d]   (sum_j attn == 1)
    #pragma unroll
    for (int nt = 0; nt < 4; ++nt) {
        const int n = nbase + nt * 16 + l16;
        float acc = 0.f;
        #pragma unroll
        for (int mt = 0; mt < 2; ++mt)
            #pragma unroll
            for (int r = 0; r < 8; ++r)
                acc += attn[h * KNB + mt * 16 + r + 8 * lhalf] * accv[nt][mt][r];
        acc += __shfl_xor(acc, 16);         // combine the two j half-sets
        if (lhalf == 0)
            out[(size_t)bn * D_DIM + n] = acc + bv[n];
    }
}

// ---------------------------------------------------------------------------
extern "C" void kernel_launch(void* const* d_in, const int* in_sizes, int n_in,
                              void* d_out, int out_size, void* d_ws, size_t ws_size,
                              hipStream_t stream) {
    (void)in_sizes; (void)n_in; (void)out_size; (void)ws_size;
    const float* neib = (const float*)d_in[0];
    const int*   mask = (const int*)  d_in[1];
    const float* x    = (const float*)d_in[2];
    const float* Wq   = (const float*)d_in[3];
    const float* bq   = (const float*)d_in[4];
    const float* Wk   = (const float*)d_in[5];
    const float* bk   = (const float*)d_in[6];
    const float* Wv   = (const float*)d_in[7];
    const float* bv   = (const float*)d_in[8];
    float* out = (float*)d_out;

    const int DD = D_DIM * D_DIM;          // 262144
    const int NX = BN_TOT * D_DIM;         // 2097152

    // Workspace layout (13.5 MiB total):
    //   [0)            Wq_bf : DD bf16
    //   [DD)           Wk_bf : DD bf16
    //   [2DD)          Wv_bf : DD bf16
    //   [3DD)          x_bf  : NX bf16
    //   then           q     : NX fp32
    bf16_t* Wq_bf = (bf16_t*)d_ws;
    bf16_t* Wk_bf = Wq_bf + DD;
    bf16_t* Wv_bf = Wk_bf + DD;
    bf16_t* x_bf  = Wv_bf + DD;
    float*  q_ws  = (float*)(x_bf + NX);   // byte offset 5,767,168 (16B aligned)

    const int cvt_total = (3 * DD + NX) / 4;   // 720,896 vec4 items (multiple of 256)
    cvt_kernel<<<cvt_total / 256, 256, 0, stream>>>(Wq, Wk, Wv, x,
                                                    Wq_bf, Wk_bf, Wv_bf, x_bf);
    q_gemm<<<BN_TOT / 32, 256, 0, stream>>>(x_bf, Wq_bf, bq, q_ws);
    gat_main<<<BN_TOT, 256, 0, stream>>>(neib, mask, Wk_bf, bk, Wv_bf, bv, q_ws, out);
}
